// SimilarAttention_44452911514112
// MI455X (gfx1250) — matrix operand order repaired and checked by reading, not compiled, not run on hardware
//
#include <hip/hip_runtime.h>
#include <stdint.h>

// Problem constants (must match reference)
#define BB   4
#define NN   2048
#define DIM  512
#define HH   8
#define DHD  64
#define INNER 512
#define QLD  2048   // qkfv row stride (4*INNER)

typedef __attribute__((ext_vector_type(16))) __bf16 v16bf;
typedef __attribute__((ext_vector_type(8)))  float  v8f;

union FragAB {
  v16bf v;
  uint32_t u[8];
  uint4 q[2];
};

__device__ __forceinline__ uint16_t f2bf(float f) {
  uint32_t u = __float_as_uint(f);
  u += 0x7FFFu + ((u >> 16) & 1u);           // round-to-nearest-even
  return (uint16_t)(u >> 16);
}

__device__ __forceinline__ v8f zero_v8f() {
  v8f z; for (int i = 0; i < 8; ++i) z[i] = 0.0f; return z;
}

// A-fragment (16x32 bf16): per lane row=idx; dwords 0..3 at K=half*8+{0,2,4,6},
// dwords 4..7 at K=16+half*8+{0,2,4,6}  -> two contiguous 16B chunks.
__device__ __forceinline__ void load_fragA(FragAB& f, const uint16_t* rowp, int half) {
  f.q[0] = *(const uint4*)(rowp + half * 8);
  f.q[1] = *(const uint4*)(rowp + 16 + half * 8);
}
// B-fragment (32x16 bf16): per lane col=idx; dwords 0..3 at K=half*8+{0,2,4,6},
// dwords 4..7 at K=16+half*8+{0,2,4,6}  -> two contiguous 16B chunks.
__device__ __forceinline__ void load_fragB(FragAB& f, const uint16_t* colp, int half) {
  f.q[0] = *(const uint4*)(colp + half * 8);
  f.q[1] = *(const uint4*)(colp + 16 + half * 8);
}
// K patterns for scalar (transposed) gathers
__device__ __forceinline__ int kpatA(int j, int half) {
  return (j < 4 ? j * 2 : 16 + (j - 4) * 2) + half * 8;
}
__device__ __forceinline__ int kpatB(int j, int half) {
  return (j < 4 ? j * 2 : 16 + (j - 4) * 2) + half * 8;
}

// ---------------- fp32 -> bf16 conversion kernels ----------------
__global__ void k_cvt(const float* __restrict__ src, uint16_t* __restrict__ dst, int n) {
  for (int i = blockIdx.x * blockDim.x + threadIdx.x; i < n; i += gridDim.x * blockDim.x)
    dst[i] = f2bf(src[i]);
}

// src [R,C] fp32 -> dst [C,R] bf16 (transpose so GEMMs read K contiguously)
__global__ void k_cvt_T(const float* __restrict__ src, uint16_t* __restrict__ dst, int R, int C) {
  int n = R * C;
  for (int i = blockIdx.x * blockDim.x + threadIdx.x; i < n; i += gridDim.x * blockDim.x) {
    int r = i / C, c = i - r * C;
    dst[(long long)c * R + r] = f2bf(src[i]);
  }
}

// ---------------- Generic bf16 WMMA GEMM: C = A[M,K] * Bt[N,K]^T ----------------
// Double-buffered LDS; 256 threads = 8 waves in a 4x2 wave grid; each wave MTxNT tiles.
// MODE 0: store bf16 C (row-major, ld=N)
// MODE 1: store fp32 C + bias[col]
// MODE 2: store bf16 C + bias[col]  (batched via blockIdx.z with strideA/strideC)
template <int MT, int NT, int MODE>
__global__ void k_gemm(const uint16_t* __restrict__ A, const uint16_t* __restrict__ Bt,
                       void* __restrict__ Cout, const float* __restrict__ bias,
                       int M, int N, int K, long long strideA, long long strideC) {
  constexpr int BM = 4 * 16 * MT;
  constexpr int BN = 2 * 16 * NT;
  constexpr int BK = 32;
  constexpr int ACH = BM * BK / 8;
  constexpr int BCH = BN * BK / 8;
  constexpr int PER = (ACH + BCH) / 256;      // uint4 chunks per thread

  __shared__ uint16_t As[2][BM * BK];
  __shared__ uint16_t Bs[2][BN * BK];

  const int tid  = threadIdx.x;
  const int lane = tid & 31, wid = tid >> 5;
  const int half = lane >> 4, idx = lane & 15;
  const int wrow = wid & 3, wcol = wid >> 2;

  const int m0 = blockIdx.y * BM;
  const int n0 = blockIdx.x * BN;
  A += (long long)blockIdx.z * strideA;

  v8f acc[MT][NT];
  for (int i = 0; i < MT; ++i)
    for (int j = 0; j < NT; ++j) acc[i][j] = zero_v8f();

  uint4 stage[PER];
  auto fetch = [&](int k0) {
    for (int i = 0; i < PER; ++i) {
      int c = tid + i * 256;
      if (c < ACH) {
        int row = c >> 2, kc = (c & 3) << 3;
        stage[i] = *(const uint4*)&A[(long long)(m0 + row) * K + k0 + kc];
      } else {
        int cc = c - ACH, row = cc >> 2, kc = (cc & 3) << 3;
        stage[i] = *(const uint4*)&Bt[(long long)(n0 + row) * K + k0 + kc];
      }
    }
  };
  auto commit = [&](int buf) {
    for (int i = 0; i < PER; ++i) {
      int c = tid + i * 256;
      if (c < ACH) {
        int row = c >> 2, kc = (c & 3) << 3;
        *(uint4*)&As[buf][row * BK + kc] = stage[i];
      } else {
        int cc = c - ACH, row = cc >> 2, kc = (cc & 3) << 3;
        *(uint4*)&Bs[buf][row * BK + kc] = stage[i];
      }
    }
  };

  fetch(0);
  commit(0);
  int cur = 0;
  for (int k0 = 0; k0 < K; k0 += BK) {
    __syncthreads();
    const bool more = (k0 + BK) < K;
    if (more) fetch(k0 + BK);            // overlap next-tile HBM fetch with WMMAs

    FragAB a[MT], b[NT];
    for (int i = 0; i < MT; ++i)
      load_fragA(a[i], &As[cur][(wrow * 16 * MT + i * 16 + idx) * BK], half);
    for (int i = 0; i < NT; ++i)
      load_fragB(b[i], &Bs[cur][(wcol * 16 * NT + i * 16 + idx) * BK], half);
    for (int i = 0; i < MT; ++i)
      for (int j = 0; j < NT; ++j)
        acc[i][j] = __builtin_amdgcn_wmma_f32_16x16x32_bf16(
            false, a[i].v, false, b[j].v, (short)0, acc[i][j], false, false);

    if (more) commit(cur ^ 1);
    cur ^= 1;
  }

  for (int i = 0; i < MT; ++i)
    for (int j = 0; j < NT; ++j) {
      int tm = m0 + wrow * 16 * MT + i * 16;
      int tn = n0 + wcol * 16 * NT + j * 16 + idx;
      for (int v = 0; v < 8; ++v) {
        int row = tm + v + 8 * half;
        float val = acc[i][j][v];
        if (MODE == 0) {
          ((uint16_t*)Cout)[(long long)row * N + tn] = f2bf(val);
        } else if (MODE == 1) {
          ((float*)Cout)[(long long)row * N + tn] = val + bias[tn];
        } else {
          ((uint16_t*)Cout)[(long long)blockIdx.z * strideC + (long long)row * N + tn] =
              f2bf(val + bias[tn]);
        }
      }
    }
}

// ---------------- poss = f^T @ v per (b,h): [64x64] = sum_n F[n,d] V[n,e] ----------------
__global__ void k_ftv(const uint16_t* __restrict__ qkfv, uint16_t* __restrict__ poss) {
  const int bh = blockIdx.x;                 // 0..31
  const int b = bh >> 3, h = bh & 7;
  __shared__ uint16_t Fs[32 * 64];
  __shared__ uint16_t Vs[32 * 64];
  const int tid  = threadIdx.x, lane = tid & 31, wid = tid >> 5;
  const int half = lane >> 4, idx = lane & 15;
  const int mt  = wid & 3;                   // d-tile 0..3
  const int ntb = (wid >> 2) * 2;            // e-tile base 0 or 2

  v8f acc[2]; acc[0] = zero_v8f(); acc[1] = zero_v8f();

  const uint16_t* Fg = qkfv + (long long)b * NN * QLD + 2 * INNER + h * DHD;
  const uint16_t* Vg = qkfv + (long long)b * NN * QLD + 3 * INNER + h * DHD;

  for (int n0 = 0; n0 < NN; n0 += 32) {
    {
      int r = tid >> 3, c = (tid & 7) << 3;  // 256 chunks each tile
      *(uint4*)&Fs[r * 64 + c] = *(const uint4*)&Fg[(long long)(n0 + r) * QLD + c];
      *(uint4*)&Vs[r * 64 + c] = *(const uint4*)&Vg[(long long)(n0 + r) * QLD + c];
    }
    __syncthreads();
    FragAB a, bf[2];
    {
      int d = mt * 16 + idx;                 // A = F^T : row=d, k=n (16-bit gathers)
      for (int j = 0; j < 8; ++j) {
        int k = kpatA(j, half);
        uint32_t lo = Fs[k * 64 + d], hi = Fs[(k + 1) * 64 + d];
        a.u[j] = lo | (hi << 16);
      }
    }
    for (int t = 0; t < 2; ++t) {
      int e = (ntb + t) * 16 + idx;          // B = V : col=e, k=n
      for (int j = 0; j < 8; ++j) {
        int k = kpatB(j, half);
        uint32_t lo = Vs[k * 64 + e], hi = Vs[(k + 1) * 64 + e];
        bf[t].u[j] = lo | (hi << 16);
      }
      acc[t] = __builtin_amdgcn_wmma_f32_16x16x32_bf16(
          false, a.v, false, bf[t].v, (short)0, acc[t], false, false);
    }
    __syncthreads();
  }
  for (int t = 0; t < 2; ++t) {
    int tn = (ntb + t) * 16 + idx;
    for (int v = 0; v < 8; ++v) {
      int row = mt * 16 + v + 8 * half;
      poss[(long long)bh * 4096 + row * 64 + tn] = f2bf(acc[t][v]);
    }
  }
}

// ------------- flash attention: out[i,d] = softmax(q k^T * s)[i,:] . pv[d,:] -------------
// 128 threads = 4 waves, 16 query rows each; K/V^T tiles double-buffered; one barrier/iter.
__global__ void k_flash(const uint16_t* __restrict__ qkfv, const uint16_t* __restrict__ pv,
                        uint16_t* __restrict__ outb) {
  const int bh = blockIdx.y;
  const int b = bh >> 3, h = bh & 7;
  const int tid = threadIdx.x, lane = tid & 31, wid = tid >> 5;
  const int half = lane >> 4, idx = lane & 15;
  const int i0 = blockIdx.x * 64 + wid * 16;

  __shared__ uint16_t Ks[2][32 * 64];      // [j][d]
  __shared__ uint16_t Vts[2][64 * 32];     // [d][j]
  __shared__ uint16_t Ps[4][16 * 32];      // per-wave P staging (wave-private)

  const uint16_t* Qg = qkfv + (long long)b * NN * QLD + h * DHD;
  const uint16_t* Kg = qkfv + (long long)b * NN * QLD + INNER + h * DHD;
  const uint16_t* Pg = pv + (long long)bh * DHD * NN;

  FragAB aq[2];                            // Q rows i0+idx, k-steps d=0..31 / 32..63
  for (int s = 0; s < 2; ++s)
    load_fragA(aq[s], Qg + (long long)(i0 + idx) * QLD + s * 32, half);

  v8f o[4]; for (int t = 0; t < 4; ++t) o[t] = zero_v8f();
  float mst[8], lst[8];
  for (int v = 0; v < 8; ++v) { mst[v] = -1e30f; lst[v] = 0.0f; }

  uint4 stK[2], stV[2];
  auto fetch = [&](int j0) {
    for (int i = 0; i < 2; ++i) {          // K tile: 256 chunks over 128 threads
      int c = tid + i * 128, r = c >> 3, cc = (c & 7) << 3;
      stK[i] = *(const uint4*)&Kg[(long long)(j0 + r) * QLD + cc];
    }
    for (int i = 0; i < 2; ++i) {          // V^T tile
      int c = tid + i * 128, r = c >> 2, cc = (c & 3) << 3;
      stV[i] = *(const uint4*)&Pg[(long long)r * NN + j0 + cc];
    }
  };
  auto commit = [&](int buf) {
    for (int i = 0; i < 2; ++i) {
      int c = tid + i * 128, r = c >> 3, cc = (c & 7) << 3;
      *(uint4*)&Ks[buf][r * 64 + cc] = stK[i];
    }
    for (int i = 0; i < 2; ++i) {
      int c = tid + i * 128, r = c >> 2, cc = (c & 3) << 3;
      *(uint4*)&Vts[buf][r * 32 + cc] = stV[i];
    }
  };

  fetch(0);
  commit(0);
  int cur = 0;
  for (int j0 = 0; j0 < NN; j0 += 32) {
    __syncthreads();
    const bool more = (j0 + 32) < NN;
    if (more) fetch(j0 + 32);

    v8f s[2]; s[0] = zero_v8f(); s[1] = zero_v8f();      // S tile 16x32
    for (int t = 0; t < 2; ++t)
      for (int ks = 0; ks < 2; ++ks) {
        FragAB bk;
        load_fragB(bk, &Ks[cur][(t * 16 + idx) * 64 + ks * 32], half);
        s[t] = __builtin_amdgcn_wmma_f32_16x16x32_bf16(
            false, aq[ks].v, false, bk.v, (short)0, s[t], false, false);
      }

    const float scl = 0.125f;                            // DH^-0.5
    float p0[8], p1[8];
    for (int v = 0; v < 8; ++v) {                        // online softmax, row = v+8*half
      float a0 = s[0][v] * scl, a1 = s[1][v] * scl;
      float rm = fmaxf(a0, a1);
      for (int msk = 1; msk < 16; msk <<= 1) rm = fmaxf(rm, __shfl_xor(rm, msk, 32));
      float mnew = fmaxf(mst[v], rm);
      float corr = __expf(mst[v] - mnew);
      float e0 = __expf(a0 - mnew), e1 = __expf(a1 - mnew);
      float rs = e0 + e1;
      for (int msk = 1; msk < 16; msk <<= 1) rs += __shfl_xor(rs, msk, 32);
      lst[v] = lst[v] * corr + rs;
      mst[v] = mnew;
      p0[v] = e0; p1[v] = e1;
      for (int t = 0; t < 4; ++t) o[t][v] = o[t][v] * corr;
    }
    // C-layout -> wave-private LDS (LDS ops are in-order per wave; no block barrier)
    for (int v = 0; v < 8; ++v) {
      int row = v + 8 * half;
      Ps[wid][row * 32 + idx]      = f2bf(p0[v]);
      Ps[wid][row * 32 + 16 + idx] = f2bf(p1[v]);
    }
    FragAB ap;
    load_fragA(ap, &Ps[wid][idx * 32], half);
    for (int t = 0; t < 4; ++t) {                        // o += P @ V
      FragAB bv;
      load_fragB(bv, &Vts[cur][(t * 16 + idx) * 32], half);
      o[t] = __builtin_amdgcn_wmma_f32_16x16x32_bf16(
          false, ap.v, false, bv.v, (short)0, o[t], false, false);
    }
    if (more) commit(cur ^ 1);
    cur ^= 1;
  }

  for (int v = 0; v < 8; ++v) {                          // normalize + scatter to [B,N,INNER]
    float inv = 1.0f / lst[v];
    int row = i0 + v + 8 * half;
    for (int t = 0; t < 4; ++t)
      outb[((long long)(b * NN + row)) * INNER + h * DHD + t * 16 + idx] =
          f2bf(o[t][v] * inv);
  }
}

// ---------------------------------------------------------------------------
extern "C" void kernel_launch(void* const* d_in, const int* in_sizes, int n_in,
                              void* d_out, int out_size, void* d_ws, size_t ws_size,
                              hipStream_t stream) {
  const float* x        = (const float*)d_in[0];
  const float* w_qkfv   = (const float*)d_in[1];
  const float* w_scales = (const float*)d_in[2];
  const float* b_scales = (const float*)d_in[3];
  const float* w_out    = (const float*)d_in[4];
  const float* b_out    = (const float*)d_in[5];

  char* ws = (char*)d_ws;
  size_t off = 0;
  auto alloc = [&](size_t bytes) {
    void* p = ws + off;
    off += (bytes + 255) & ~(size_t)255;
    return p;
  };
  uint16_t* xb     = (uint16_t*)alloc(8192ull * 512 * 2);     // x bf16
  uint16_t* wqkfvT = (uint16_t*)alloc(2048ull * 512 * 2);     // w_qkfv^T bf16
  uint16_t* woutT  = (uint16_t*)alloc(512ull * 512 * 2);      // w_out^T bf16
  uint16_t* wsclT  = (uint16_t*)alloc(2048ull * 64 * 2);      // w_scales^T bf16
  uint16_t* qkfvb  = (uint16_t*)alloc(8192ull * 2048 * 2);    // qkfv bf16 [B*N, 4*INNER]
  uint16_t* possb  = (uint16_t*)alloc(32ull * 64 * 64 * 2);   // f^T v per (b,h)
  uint16_t* pvb    = (uint16_t*)alloc(32ull * 64 * 2048 * 2); // poss_value [bh][d][n]
  uint16_t* aoutb  = (uint16_t*)alloc(8192ull * 512 * 2);     // attn out [B*N, INNER]

  k_cvt  <<<1024, 256, 0, stream>>>(x, xb, 8192 * 512);
  k_cvt_T<<<1024, 256, 0, stream>>>(w_qkfv, wqkfvT, 512, 2048);
  k_cvt_T<<< 256, 256, 0, stream>>>(w_out, woutT, 512, 512);
  k_cvt_T<<< 128, 256, 0, stream>>>(w_scales, wsclT, 64, 2048);

  // qkfv = x @ w_qkfv   [8192,512]x[512,2048] -> bf16
  k_gemm<2, 4, 0><<<dim3(2048 / 128, 8192 / 128, 1), 256, 0, stream>>>(
      xb, wqkfvT, qkfvb, nullptr, 8192, 2048, 512, 0, 0);

  // poss = f^T @ v per head
  k_ftv<<<32, 256, 0, stream>>>(qkfvb, possb);

  // pv = poss @ w_scales + b_scales  (batched over 32 heads)
  k_gemm<1, 4, 2><<<dim3(2048 / 128, 1, 32), 256, 0, stream>>>(
      possb, wsclT, pvb, b_scales, 64, 2048, 64, 64ll * 64, 64ll * 2048);

  // flash attention: softmax(q k^T * s) @ pv^T
  k_flash<<<dim3(2048 / 64, 32, 1), 128, 0, stream>>>(qkfvb, pvb, aoutb);

  // out = attn_out @ w_out + b_out  (fp32 result)
  k_gemm<2, 4, 1><<<dim3(512 / 128, 8192 / 128, 1), 256, 0, stream>>>(
      aoutb, woutT, (float*)d_out, b_out, 8192, 512, 512, 0, 0);
}
